// CaptioningRNN_12824772346402
// MI455X (gfx1250) — matrix operand order repaired and checked by reading, not compiled, not run on hardware
//
#include <hip/hip_runtime.h>
#include <hip/hip_bf16.h>
#include <stdint.h>

// ---------------------------------------------------------------------------
// Types for WMMA fragments (gfx1250, wave32)
// ---------------------------------------------------------------------------
typedef __attribute__((ext_vector_type(16))) __bf16 v16bf;
typedef __attribute__((ext_vector_type(8)))  float  v8f;
typedef __attribute__((ext_vector_type(4)))  unsigned int u32x4;

union FragU { u32x4 u[2]; v16bf v; };

// float -> bf16 bits, round-to-nearest-even
__device__ __forceinline__ unsigned short f2bf(float f) {
    union { float f; unsigned int u; } cv; cv.f = f;
    unsigned int u = cv.u;
    unsigned int r = (u + 0x7FFFu + ((u >> 16) & 1u)) >> 16;
    return (unsigned short)r;
}

// A-matrix fragment, 16x32 bf16 (rows row0..row0+15, K = k..k+31).
// ISA layout: lane l holds row (l&15); VGPR0-3: K = koff+0..7, VGPR4-7: K = 16+koff+0..7,
// koff = 8*(l>>4). Each half is 8 contiguous bf16 = one 16B load.
__device__ __forceinline__ v16bf load_a_frag(const unsigned short* base, size_t stride,
                                             int row0, int k, int lane) {
    int r = lane & 15;
    int koff = (lane >> 4) << 3;
    const unsigned short* p = base + (size_t)(row0 + r) * stride + (size_t)(k + koff);
    FragU f;
    f.u[0] = *reinterpret_cast<const u32x4*>(p);
    f.u[1] = *reinterpret_cast<const u32x4*>(p + 16);
    return f.v;
}

// B-matrix fragment, 32x16 bf16 (K = k..k+31, cols col0..col0+15), weights stored
// transposed as (col, K) row-major so each lane reads two runs of 8 contiguous bf16.
// Layout: lane l holds col (l&15); VGPR0-3: K = koff+0..7, VGPR4-7: K = 16+koff+0..7,
// koff = 8*(l>>4).
__device__ __forceinline__ v16bf load_b_frag(const unsigned short* baseT, size_t stride,
                                             int col0, int k, int lane) {
    int c = lane & 15;
    int koff = (lane >> 4) << 3;
    const unsigned short* p = baseT + (size_t)(col0 + c) * stride + (size_t)(k + koff);
    FragU f;
    f.u[0] = *reinterpret_cast<const u32x4*>(p);
    f.u[1] = *reinterpret_cast<const u32x4*>(p + 16);
    return f.v;
}

// ---------------------------------------------------------------------------
// Prep: transpose + convert one 1024x4096 fp32 weight into 4096x1024 bf16
// WT[c*1024 + k] = bf16(W[k*4096 + c])
// ---------------------------------------------------------------------------
__global__ __launch_bounds__(256) void wconv_kernel(const float* __restrict__ W,
                                                    unsigned short* __restrict__ WT) {
    int idx = blockIdx.x * 256 + threadIdx.x;      // over output, 0 .. 4096*1024-1
    int c = idx >> 10;
    int k = idx & 1023;
    WT[idx] = f2bf(W[(size_t)k * 4096 + c]);
}

// Prep: convert x (N*T*D fp32) to bf16
__global__ __launch_bounds__(256) void xconv_kernel(const float* __restrict__ x,
                                                    unsigned short* __restrict__ xbf) {
    int idx = blockIdx.x * 256 + threadIdx.x;
    xbf[idx] = f2bf(x[idx]);
}

// ---------------------------------------------------------------------------
// Shared attention helper: given this thread's 4 h values (j = tid + 256*i),
// compute scores = (h . A_flat)/sqrt(H), 16-wide softmax, attn; pack h and attn
// to bf16 for the next gate GEMM. Block = one batch element, 256 threads.
// ---------------------------------------------------------------------------
__device__ __forceinline__ void attention_and_pack(
    int n, int tid, const float* __restrict__ An, const float hv[4],
    unsigned short* __restrict__ hbf, unsigned short* __restrict__ attnbf,
    float* s_part /*256*16*/, float* s_scores /*16*/) {

    float sc[16];
#pragma unroll
    for (int p = 0; p < 16; ++p) sc[p] = 0.f;
#pragma unroll
    for (int i = 0; i < 4; ++i) {
        int j = tid + i * 256;
        const float* Aj = An + (size_t)j * 16;
#pragma unroll
        for (int p = 0; p < 16; ++p) sc[p] += hv[i] * Aj[p];
    }
#pragma unroll
    for (int p = 0; p < 16; ++p) s_part[tid * 16 + p] = sc[p];
    __syncthreads();

    if (tid < 16) {
        float acc = 0.f;
        for (int i = 0; i < 256; ++i) acc += s_part[i * 16 + tid];
        s_scores[tid] = acc * 0.03125f;            // 1/sqrt(1024)
    }
    __syncthreads();

    // every thread computes the 16-wide softmax redundantly (cheap)
    float scl[16];
    float m = -3.0e38f;
#pragma unroll
    for (int p = 0; p < 16; ++p) { scl[p] = s_scores[p]; m = fmaxf(m, scl[p]); }
    float sum = 0.f;
    float w[16];
#pragma unroll
    for (int p = 0; p < 16; ++p) { w[p] = __expf(scl[p] - m); sum += w[p]; }
    float inv = 1.0f / sum;

#pragma unroll
    for (int i = 0; i < 4; ++i) {
        int j = tid + i * 256;
        const float* Aj = An + (size_t)j * 16;
        float a = 0.f;
#pragma unroll
        for (int p = 0; p < 16; ++p) a += Aj[p] * w[p];
        attnbf[(size_t)n * 1024 + j] = f2bf(a * inv);
        hbf[(size_t)n * 1024 + j] = f2bf(hv[i]);
    }
}

// ---------------------------------------------------------------------------
// Init: h0 = c0 = mean over 16 spatial positions; then attention for t=0.
// ---------------------------------------------------------------------------
__global__ __launch_bounds__(256) void init_kernel(const float* __restrict__ A,
                                                   float* __restrict__ c,
                                                   unsigned short* __restrict__ hbf,
                                                   unsigned short* __restrict__ attnbf) {
    __shared__ float s_part[256 * 16];
    __shared__ float s_scores[16];
    const int n = blockIdx.x;
    const int tid = threadIdx.x;
    const float* An = A + (size_t)n * 1024 * 16;

    float hv[4];
#pragma unroll
    for (int i = 0; i < 4; ++i) {
        int j = tid + i * 256;
        const float* Aj = An + (size_t)j * 16;
        float s = 0.f;
#pragma unroll
        for (int p = 0; p < 16; ++p) s += Aj[p];
        hv[i] = s * (1.0f / 16.0f);
        c[(size_t)n * 1024 + j] = hv[i];
    }
    attention_and_pack(n, tid, An, hv, hbf, attnbf, s_part, s_scores);
}

// ---------------------------------------------------------------------------
// Fused LSTM pointwise update + next-step attention. h stays in registers from
// the gate update through attention packing (never hits fp32 global memory).
// Writes out[:, t, :], updates c, and produces hbf/attnbf for step t+1.
// ---------------------------------------------------------------------------
__global__ __launch_bounds__(256) void update_attn_kernel(
    const float* __restrict__ a,       // (N, 4096) gate pre-activations
    const float* __restrict__ A,       // (N, H, 16)
    float* __restrict__ c,             // (N, H)
    float* __restrict__ out,           // (N, T, H)
    unsigned short* __restrict__ hbf,
    unsigned short* __restrict__ attnbf,
    int t)
{
    __shared__ float s_part[256 * 16];
    __shared__ float s_scores[16];
    const int n = blockIdx.x;
    const int tid = threadIdx.x;
    const float* An = A + (size_t)n * 1024 * 16;
    const float* arow = a + (size_t)n * 4096;

    float hv[4];
#pragma unroll
    for (int i = 0; i < 4; ++i) {
        int j = tid + i * 256;
        float ai = arow[j];
        float af = arow[1024 + j];
        float ao = arow[2048 + j];
        float ag = arow[3072 + j];
        float cp = c[(size_t)n * 1024 + j];
        float si = 1.0f / (1.0f + __expf(-ai));
        float sf = 1.0f / (1.0f + __expf(-af));
        float so = 1.0f / (1.0f + __expf(-ao));
        float tg = tanhf(ag);
        float nc = sf * cp + si * tg;
        float nh = so * tanhf(nc);
        c[(size_t)n * 1024 + j] = nc;
        out[((size_t)n * 64 + t) * 1024 + j] = nh;
        hv[i] = nh;
    }
    // prepare bf16 h / attn for step t+1 (harmless extra work at t = T-1)
    attention_and_pack(n, tid, An, hv, hbf, attnbf, s_part, s_scores);
}

// ---------------------------------------------------------------------------
// Fused gate GEMM: a = x_t @ Wx + h @ Wh + attn @ Wattn + b   -> (N, 4096) fp32
// Each wave computes a 32(N) x 64(4H) tile: 8 f32 accumulators, 24 WMMAs per
// K-step, 2 WMMAs per B fragment (halves weight traffic vs 16-row tiles).
// Grid: 64 blocks x 8 waves = 512 tiles (8 row-tiles x 64 col-tiles).
// ---------------------------------------------------------------------------
__global__ __launch_bounds__(256) void gate_gemm_kernel(
    const unsigned short* __restrict__ xbf,    // (N, T, D) bf16
    const unsigned short* __restrict__ hbf,    // (N, H) bf16
    const unsigned short* __restrict__ attnbf, // (N, H) bf16
    const unsigned short* __restrict__ WxT,    // (4H, D) bf16 (transposed)
    const unsigned short* __restrict__ WhT,    // (4H, H) bf16
    const unsigned short* __restrict__ WattnT, // (4H, H) bf16
    const float* __restrict__ bias,            // (4H)
    float* __restrict__ a_out,                 // (N, 4H) fp32
    int t)
{
    const int lane = threadIdx.x & 31;
    const int wave = blockIdx.x * 8 + (threadIdx.x >> 5);   // 0..511
    const int rt = wave >> 6;                               // 0..7
    const int ct = wave & 63;                               // 0..63
    const int row0 = rt * 32;                               // batch rows
    const int col0 = ct * 64;                               // gate columns

    v8f acc[8] = {};   // [0..3]: rows row0..+15, [4..7]: rows row0+16..+31

    // x_t rows: element (n, k) at ((n*64 + t)*1024 + k)
    const unsigned short* xrow = xbf + (size_t)t * 1024;

    for (int k = 0; k < 1024; k += 32) {
        v16bf ax0 = load_a_frag(xrow,   65536, row0,      k, lane);
        v16bf ax1 = load_a_frag(xrow,   65536, row0 + 16, k, lane);
        v16bf ah0 = load_a_frag(hbf,    1024,  row0,      k, lane);
        v16bf ah1 = load_a_frag(hbf,    1024,  row0 + 16, k, lane);
        v16bf aa0 = load_a_frag(attnbf, 1024,  row0,      k, lane);
        v16bf aa1 = load_a_frag(attnbf, 1024,  row0 + 16, k, lane);
#pragma unroll
        for (int cblk = 0; cblk < 4; ++cblk) {
            int col = col0 + cblk * 16;
            v16bf bx = load_b_frag(WxT, 1024, col, k, lane);
            acc[cblk] = __builtin_amdgcn_wmma_f32_16x16x32_bf16(
                false, ax0, false, bx, (short)0, acc[cblk], false, false);
            acc[4 + cblk] = __builtin_amdgcn_wmma_f32_16x16x32_bf16(
                false, ax1, false, bx, (short)0, acc[4 + cblk], false, false);
            v16bf bh = load_b_frag(WhT, 1024, col, k, lane);
            acc[cblk] = __builtin_amdgcn_wmma_f32_16x16x32_bf16(
                false, ah0, false, bh, (short)0, acc[cblk], false, false);
            acc[4 + cblk] = __builtin_amdgcn_wmma_f32_16x16x32_bf16(
                false, ah1, false, bh, (short)0, acc[4 + cblk], false, false);
            v16bf ba = load_b_frag(WattnT, 1024, col, k, lane);
            acc[cblk] = __builtin_amdgcn_wmma_f32_16x16x32_bf16(
                false, aa0, false, ba, (short)0, acc[cblk], false, false);
            acc[4 + cblk] = __builtin_amdgcn_wmma_f32_16x16x32_bf16(
                false, aa1, false, ba, (short)0, acc[4 + cblk], false, false);
        }
    }

    // C/D layout: lane l -> col = l&15; VGPR r -> row = r + 8*(l>>4)
    const int colLane = lane & 15;
    const int rbase = (lane >> 4) * 8;
#pragma unroll
    for (int cblk = 0; cblk < 4; ++cblk) {
        int col = col0 + cblk * 16 + colLane;
        float bv = bias[col];
#pragma unroll
        for (int r = 0; r < 8; ++r) {
            a_out[(size_t)(row0 + rbase + r) * 4096 + col]      = acc[cblk][r] + bv;
            a_out[(size_t)(row0 + 16 + rbase + r) * 4096 + col] = acc[4 + cblk][r] + bv;
        }
    }
}

// ---------------------------------------------------------------------------
// Host side
// ---------------------------------------------------------------------------
extern "C" void kernel_launch(void* const* d_in, const int* in_sizes, int n_in,
                              void* d_out, int out_size, void* d_ws, size_t ws_size,
                              hipStream_t stream) {
    (void)in_sizes; (void)n_in; (void)out_size; (void)ws_size;

    const float* x     = (const float*)d_in[0];   // (256, 64, 1024)
    const float* A     = (const float*)d_in[1];   // (256, 1024, 4, 4)
    const float* Wx    = (const float*)d_in[2];   // (1024, 4096)
    const float* Wh    = (const float*)d_in[3];   // (1024, 4096)
    const float* Wattn = (const float*)d_in[4];   // (1024, 4096)
    const float* b     = (const float*)d_in[5];   // (4096)
    float* out = (float*)d_out;                   // (256, 64, 1024)

    char* ws = (char*)d_ws;
    auto carve = [&](size_t bytes) -> char* {
        char* p = ws;
        ws += (bytes + 255) & ~(size_t)255;
        return p;
    };
    unsigned short* WxT    = (unsigned short*)carve((size_t)4096 * 1024 * 2);
    unsigned short* WhT    = (unsigned short*)carve((size_t)4096 * 1024 * 2);
    unsigned short* WattnT = (unsigned short*)carve((size_t)4096 * 1024 * 2);
    unsigned short* xbf    = (unsigned short*)carve((size_t)256 * 64 * 1024 * 2);
    unsigned short* hbf    = (unsigned short*)carve((size_t)256 * 1024 * 2);
    unsigned short* attnbf = (unsigned short*)carve((size_t)256 * 1024 * 2);
    float* cbuf = (float*)carve((size_t)256 * 1024 * 4);
    float* abuf = (float*)carve((size_t)256 * 4096 * 4);

    // Prep: weight transpose+convert, x convert, initial state + t=0 attention
    wconv_kernel<<<(4096 * 1024) / 256, 256, 0, stream>>>(Wx, WxT);
    wconv_kernel<<<(4096 * 1024) / 256, 256, 0, stream>>>(Wh, WhT);
    wconv_kernel<<<(4096 * 1024) / 256, 256, 0, stream>>>(Wattn, WattnT);
    xconv_kernel<<<(256 * 64 * 1024) / 256, 256, 0, stream>>>(x, xbf);
    init_kernel<<<256, 256, 0, stream>>>(A, cbuf, hbf, attnbf);

    // Recurrent loop: 64 sequential steps, 2 launches per step
    for (int t = 0; t < 64; ++t) {
        gate_gemm_kernel<<<64, 256, 0, stream>>>(xbf, hbf, attnbf,
                                                 WxT, WhT, WattnT, b, abuf, t);
        update_attn_kernel<<<256, 256, 0, stream>>>(abuf, A, cbuf, out, hbf, attnbf, t);
    }
}
